// Cortex_28114855919744
// MI455X (gfx1250) — matrix operand order repaired and checked by reading, not compiled, not run on hardware
//
#include <hip/hip_runtime.h>
#include <hip/hip_bf16.h>

typedef unsigned int u32;
typedef __attribute__((ext_vector_type(16))) __bf16 v16bf;
typedef __attribute__((ext_vector_type(4)))  __bf16 v4bf;
typedef __attribute__((ext_vector_type(8)))  float  v8f;

#define TILE_M 128
#define TILE_N 128
#define TILE_K 32
#define LDS_ROW 20   // u32 per LDS row = 40 bf16 (32 data + 8 pad) -> balanced b128 reads

#if __has_builtin(__builtin_amdgcn_sched_barrier)
#define SCHED_FENCE() __builtin_amdgcn_sched_barrier(0)
#else
#define SCHED_FENCE()
#endif

// native RNE converts -> v_cvt_pk_bf16_f32 on gfx1250
__device__ __forceinline__ uint2 pack_bf16_4(float4 f) {
  v4bf v;
  v.x = (__bf16)f.x; v.y = (__bf16)f.y; v.z = (__bf16)f.z; v.w = (__bf16)f.w;
  union { v4bf v; uint2 u; } cv;
  cv.v = v;
  return cv.u;
}

// C[M,N] = A[M,K] (f32, row-major, lda) @ W[N,K]^T (f32, row-major, ldw) + bias
// mode 0: store; mode 1: exact gelu then store; mode 2: add resid then store
__global__ __launch_bounds__(256)
void gemm_bf16_kernel(const float* __restrict__ A, int lda,
                      const float* __restrict__ W, int ldw,
                      const float* __restrict__ bias,
                      float* __restrict__ C, int ldc,
                      const float* __restrict__ resid,
                      int K, int mode) {
  __shared__ __align__(16) u32 sA[2][TILE_M * LDS_ROW];
  __shared__ __align__(16) u32 sB[2][TILE_N * LDS_ROW];

  const int t  = threadIdx.x;
  const int m0 = blockIdx.y * TILE_M;
  const int n0 = blockIdx.x * TILE_N;

  const int lane = t & 31;
  const int wv   = t >> 5;
  const int wm   = (wv >> 2) * 64;   // wave M offset (2 wave-rows)
  const int wn   = (wv & 3) * 32;    // wave N offset (4 wave-cols)
  const int lg   = lane >> 4;        // lane group 0/1
  const int lr   = lane & 15;

  v8f acc[4][2];
  #pragma unroll
  for (int mi = 0; mi < 4; ++mi)
    #pragma unroll
    for (int ni = 0; ni < 2; ++ni)
      #pragma unroll
      for (int r = 0; r < 8; ++r) acc[mi][ni][r] = 0.0f;

  const int nk = K / TILE_K;   // >= 16 for all GEMMs in this pipeline
  float4 ra[4], rb[4];

  auto load_tile = [&](int kt) {
    #pragma unroll
    for (int i = 0; i < 4; ++i) {
      int idx = t + i * 256;          // 0..1023 float4 slots
      int row = idx >> 3;             // 0..127
      int c4  = idx & 7;              // float4 column within 32-wide K tile
      ra[i] = *(const float4*)(A + (size_t)(m0 + row) * lda + kt * TILE_K + c4 * 4);
      rb[i] = *(const float4*)(W + (size_t)(n0 + row) * ldw + kt * TILE_K + c4 * 4);
    }
  };
  auto store_tile = [&](int buf) {
    #pragma unroll
    for (int i = 0; i < 4; ++i) {
      int idx = t + i * 256;
      int row = idx >> 3;
      int c4  = idx & 7;
      *(uint2*)&sA[buf][row * LDS_ROW + c4 * 2] = pack_bf16_4(ra[i]);
      *(uint2*)&sB[buf][row * LDS_ROW + c4 * 2] = pack_bf16_4(rb[i]);
    }
  };
  auto compute = [&](int cb) {
    union Frag { v16bf v; uint4 q[2]; } afr[4], bfr[2];
    // A fragment: lane row m = wm+mi*16+lr; K chunks {8lg..8lg+7} and {16+8lg..16+8lg+7}
    #pragma unroll
    for (int mi = 0; mi < 4; ++mi) {
      const u32* base = &sA[cb][(wm + mi * 16 + lr) * LDS_ROW];
      afr[mi].q[0] = *(const uint4*)(base + 4 * lg);
      afr[mi].q[1] = *(const uint4*)(base + 8 + 4 * lg);
    }
    // B fragment: lane col n = wn+ni*16+lr; K chunks {8lg..8lg+7} and {16+8lg..16+8lg+7}
    #pragma unroll
    for (int ni = 0; ni < 2; ++ni) {
      const u32* base = &sB[cb][(wn + ni * 16 + lr) * LDS_ROW];
      bfr[ni].q[0] = *(const uint4*)(base + 4 * lg);
      bfr[ni].q[1] = *(const uint4*)(base + 8 + 4 * lg);
    }
    #pragma unroll
    for (int mi = 0; mi < 4; ++mi)
      #pragma unroll
      for (int ni = 0; ni < 2; ++ni)
        acc[mi][ni] = __builtin_amdgcn_wmma_f32_16x16x32_bf16(
            false, afr[mi].v, false, bfr[ni].v, (short)0, acc[mi][ni], false, false);
  };

  load_tile(0);
  store_tile(0);
  __syncthreads();

  // steady state, phase-fenced software pipeline:
  //   [8x global_load_b128] | [ds_load + wmma burst] | [wait once, cvt, ds_store] | barrier
  for (int kt = 0; kt < nk - 1; ++kt) {
    load_tile(kt + 1);
    SCHED_FENCE();            // keep loads issued back-to-back, no early cvt/wait
    compute(kt & 1);
    SCHED_FENCE();            // conversions/stores stay after the WMMA burst
    store_tile((kt + 1) & 1);
    __syncthreads();
  }
  compute((nk - 1) & 1);

  // Epilogue. C/D layout: VGPR r -> M = r + 8*lg, N = lr.
  #pragma unroll
  for (int ni = 0; ni < 2; ++ni) {
    const int col = n0 + wn + ni * 16 + lr;
    const float bv = bias ? bias[col] : 0.0f;
    #pragma unroll
    for (int mi = 0; mi < 4; ++mi) {
      const int rbase = m0 + wm + mi * 16 + 8 * lg;
      #pragma unroll
      for (int r = 0; r < 8; ++r) {
        float v = acc[mi][ni][r] + bv;
        const size_t off = (size_t)(rbase + r) * ldc + col;
        if (mode == 1)      v = 0.5f * v * (1.0f + erff(v * 0.70710678118654752f));
        else if (mode == 2) v += resid[off];
        C[off] = v;
      }
    }
  }
}

// ---- row-wise helpers over width 512 (one block per row, 256 threads) ----

__device__ __forceinline__ float2 stats512(float a0, float a1, float2* sred) {
  const int t = threadIdx.x;
  float2 p; p.x = a0 + a1; p.y = a0 * a0 + a1 * a1;
  sred[t] = p; __syncthreads();
  for (int s = 128; s > 0; s >>= 1) {
    if (t < s) { sred[t].x += sred[t + s].x; sred[t].y += sred[t + s].y; }
    __syncthreads();
  }
  float mean = sred[0].x * (1.0f / 512.0f);
  float rstd = rsqrtf(sred[0].y * (1.0f / 512.0f) - mean * mean + 1e-5f);
  __syncthreads();
  float2 r; r.x = mean; r.y = rstd; return r;
}

__global__ __launch_bounds__(256)
void ln_kernel(const float* __restrict__ x, const float* __restrict__ g,
               const float* __restrict__ b, float* __restrict__ y) {
  __shared__ float2 sred[256];
  const int row = blockIdx.x, t = threadIdx.x;
  const size_t o0 = (size_t)row * 512 + t, o1 = o0 + 256;
  const float v0 = x[o0], v1 = x[o1];
  float2 s = stats512(v0, v1, sred);
  y[o0] = (v0 - s.x) * s.y * g[t]       + b[t];
  y[o1] = (v1 - s.x) * s.y * g[t + 256] + b[t + 256];
}

// acts = LN(acts + 0.5*(LN(up)-tgt)*sigmoid(plog) + 0.1*lat)   (in-place; tgt may alias acts)
__global__ __launch_bounds__(256)
void settle_kernel(const float* __restrict__ up, const float* __restrict__ lat,
                   const float* __restrict__ tgt, float* __restrict__ acts,
                   const float* __restrict__ g, const float* __restrict__ b,
                   const float* __restrict__ plog) {
  __shared__ float2 sred[256];
  const int row = blockIdx.x, t = threadIdx.x;
  const size_t o0 = (size_t)row * 512 + t, o1 = o0 + 256;
  const float u0 = up[o0], u1 = up[o1];
  float2 s = stats512(u0, u1, sred);
  const float c0 = (u0 - s.x) * s.y * g[t]       + b[t];
  const float c1 = (u1 - s.x) * s.y * g[t + 256] + b[t + 256];
  const float p0 = 1.0f / (1.0f + expf(-plog[t]));
  const float p1 = 1.0f / (1.0f + expf(-plog[t + 256]));
  const float n0 = acts[o0] + 0.5f * (c0 - tgt[o0]) * p0 + 0.1f * lat[o0];
  const float n1 = acts[o1] + 0.5f * (c1 - tgt[o1]) * p1 + 0.1f * lat[o1];
  float2 s2 = stats512(n0, n1, sred);
  acts[o0] = (n0 - s2.x) * s2.y * g[t]       + b[t];
  acts[o1] = (n1 - s2.x) * s2.y * g[t + 256] + b[t + 256];
}

extern "C" void kernel_launch(void* const* d_in, const int* in_sizes, int n_in,
                              void* d_out, int out_size, void* d_ws, size_t ws_size,
                              hipStream_t stream) {
  (void)in_sizes; (void)n_in; (void)out_size; (void)ws_size;
  const int NTOK = 8192;             // B*S
  const int DM = 2048, DC = 512;

  const float* qwen  = (const float*)d_in[0];
  const float* obs   = (const float*)d_in[1];
  const float* projW = (const float*)d_in[2];
  const float* projb = (const float*)d_in[3];
  const float* fuseW = (const float*)d_in[4];
  const float* fuseb = (const float*)d_in[5];
  const float* upW   = (const float*)d_in[6];
  const float* upb   = (const float*)d_in[7];
  const float* latW  = (const float*)d_in[8];
  const float* latb  = (const float*)d_in[9];
  const float* plog  = (const float*)d_in[10];
  const float* lng   = (const float*)d_in[11];
  const float* lnb   = (const float*)d_in[12];
  const float* dwnW  = (const float*)d_in[13];
  const float* dwnb  = (const float*)d_in[14];
  const float* o1W   = (const float*)d_in[15];
  const float* o1b   = (const float*)d_in[16];
  const float* o2W   = (const float*)d_in[17];
  const float* o2b   = (const float*)d_in[18];

  float* ws = (float*)d_ws;
  const size_t SLOT = (size_t)NTOK * DC;        // 4,194,304 floats
  float* cat   = ws;                             // NTOK*DM = 4*SLOT (start only)
  float* upo   = ws;                             // reuse cat region after fuse
  float* lato  = ws + SLOT;
  float* pred0 = ws + 2 * SLOT;
  float* pred1 = ws + 3 * SLOT;
  float* x0    = ws + 4 * SLOT;
  float* acts  = ws + 5 * SLOT;                  // 4 consecutive SLOTs
  float* h     = upo;                            // reuse after settling

  auto gemm = [&](const float* Ap, int K, const float* Wp, const float* bp,
                  float* Cp, int ldc, int N, int mode, const float* rp) {
    dim3 grid(N / TILE_N, NTOK / TILE_M);
    gemm_bf16_kernel<<<grid, dim3(256), 0, stream>>>(Ap, K, Wp, K, bp, Cp, ldc, rp, K, mode);
  };

  // 1) proj: 4 GEMMs into concatenated columns of cat (ldc = 2048)
  for (int o = 0; o < 4; ++o)
    gemm(obs + (size_t)o * NTOK * DM, DM, projW + (size_t)o * DC * DM,
         projb + o * DC, cat + o * DC, DM, DC, 0, nullptr);

  // 2) fuse: x0 = cat @ fuseW^T + fuseb
  gemm(cat, DM, fuseW, fuseb, x0, DC, DC, 0, nullptr);

  // 3) initial bottom-up sweep
  const float* x = x0;
  for (int i = 0; i < 4; ++i) {
    gemm(x, DC, upW + (size_t)i * DC * DC, upb + i * DC, upo, DC, DC, 0, nullptr);
    ln_kernel<<<NTOK, 256, 0, stream>>>(upo, lng + i * DC, lnb + i * DC, acts + i * SLOT);
    x = acts + i * SLOT;
  }

  // 4) settling iterations
  for (int it = 0; it < 5; ++it) {
    gemm(acts + 2 * SLOT, DC, dwnW + 2 * (size_t)DC * DC, dwnb + 2 * DC, pred1, DC, DC, 0, nullptr);
    gemm(acts + 1 * SLOT, DC, dwnW + 1 * (size_t)DC * DC, dwnb + 1 * DC, pred0, DC, DC, 0, nullptr);
    const float* tgt[4] = { pred0, pred1, acts + 2 * SLOT, acts + 3 * SLOT };
    const float* xi = x0;
    for (int i = 0; i < 4; ++i) {
      gemm(xi, DC, upW + (size_t)i * DC * DC, upb + i * DC, upo, DC, DC, 0, nullptr);
      gemm(acts + i * SLOT, DC, latW + (size_t)i * DC * DC, latb + i * DC, lato, DC, DC, 0, nullptr);
      settle_kernel<<<NTOK, 256, 0, stream>>>(upo, lato, tgt[i], acts + i * SLOT,
                                              lng + i * DC, lnb + i * DC, plog + i * DC);
      xi = acts + i * SLOT;
    }
  }

  // 5) output head: h = gelu(acts3 @ o1W^T + b); out = qwen + h @ o2W^T + b
  gemm(acts + 3 * SLOT, DC, o1W, o1b, h, DC, DC, 1, nullptr);
  gemm(h, DC, o2W, o2b, (float*)d_out, DM, DM, 2, qwen);
}
